// ScaledDotProductAttention_67997922230646
// MI455X (gfx1250) — compile-verified, matrix operand order checked by reading, not run on hardware
//
#include <hip/hip_runtime.h>
#include <math.h>

typedef _Float16 half_t;
typedef __attribute__((ext_vector_type(16))) _Float16 v16h;
typedef __attribute__((ext_vector_type(8)))  _Float16 h8;
typedef __attribute__((ext_vector_type(2)))  _Float16 h2;
typedef __attribute__((ext_vector_type(8)))  float    v8f;

#define SEQ   2048
#define DH    64
#define NBH   64            // B*H = 4*16
#define KB    32            // keys per block
#define QW    16            // query rows per wave
#define NWAVE 8
#define QWG   (QW * NWAVE)  // 128 query rows per workgroup
#define KSTR  72            // LDS row stride (halves) for K  [key][d]
#define VSTR  36            // LDS row stride (halves) for V^T [d][key]
// 1/sqrt(64) * log2(e): softmax runs in the log2 domain (v_exp_f32 is exp2)
#define SCALE (0.125f * 1.44269504088896340736f)

// SWAPX16: ds_swizzle group-of-32, xor=0x10, or=0, and=0x1f -> offset 0x401F
__device__ __forceinline__ float swap_half(float x) {
    return __int_as_float(__builtin_amdgcn_ds_swizzle(__float_as_int(x), 0x401F));
}

// ---- per-block flash step; MASK only for the single causal-boundary block ----
template<bool MASK>
__device__ __forceinline__
void fa_block(const half_t* __restrict__ kTile, const half_t* __restrict__ vTile,
              int kb, int qrow, int ln16, int hh,
              const v16h bq[2], const v16h& ones,
              v8f oacc[4], v8f& lacc, float& m)
{
    // ---- S^T = K * Q^T : two 16x16 C tiles (keys x queries) ----
    v8f st[2];
    #pragma unroll
    for (int t = 0; t < 2; ++t) {
        v8f acc = (v8f){};
        #pragma unroll
        for (int c = 0; c < 2; ++c) {
            v16h a;  // A-fragment: K rows (M=key, K=d)
            const half_t* kr = &kTile[(t * 16 + ln16) * KSTR + c * 32 + hh * 8];
            #pragma unroll
            for (int v = 0; v < 4; ++v) {
                h2 p0 = *(const h2*)(kr + 2 * v);
                h2 p1 = *(const h2*)(kr + 2 * v + 16);
                a[2 * v]     = p0[0];
                a[2 * v + 1] = p0[1];
                a[2 * v + 8] = p1[0];
                a[2 * v + 9] = p1[1];
            }
            acc = __builtin_amdgcn_wmma_f32_16x16x32_f16(
                      false, a, false, bq[c], (short)0, acc, false, false);
        }
        st[t] = acc;
    }

    // ---- causal mask only on the boundary block ----
    float s[16];
    #pragma unroll
    for (int t = 0; t < 2; ++t)
        #pragma unroll
        for (int r = 0; r < 8; ++r) {
            float val = st[t][r];
            if (MASK) {
                const int key = kb + t * 16 + r + 8 * hh;
                if (key > qrow) val = -__builtin_inff();
            }
            s[t * 8 + r] = val;
        }

    // ---- online softmax in log2 domain (per-query = per-lane column) ----
    float mx = s[0];
    #pragma unroll
    for (int i = 1; i < 16; ++i) mx = fmaxf(mx, s[i]);
    mx = fmaxf(mx, swap_half(mx));            // other 16 keys live in lane^16
    const float mnew = fmaxf(m, mx);

    float p[16];
    #pragma unroll
    for (int i = 0; i < 16; ++i)
        p[i] = __builtin_amdgcn_exp2f(s[i] - mnew);  // raw v_exp_f32 (exp2)

    const float alpha = __builtin_amdgcn_exp2f(m - mnew);  // exp2(-inf)=0: first block
    m = mnew;

    // ---- P^T (C-layout) is exactly the B-fragment for O^T = V^T * P^T ----
    v16h bp;
    #pragma unroll
    for (int i = 0; i < 8; ++i) {
        auto pk = __builtin_amdgcn_cvt_pkrtz(p[2 * i], p[2 * i + 1]);
        bp[2 * i]     = (_Float16)pk[0];
        bp[2 * i + 1] = (_Float16)pk[1];
    }

    // ---- l accumulated on the matrix pipe: ones(16x32) x P^T = rowsum(P) ----
    {
        v8f lc = lacc;
        #pragma unroll
        for (int r = 0; r < 8; ++r) lc[r] *= alpha;
        lacc = __builtin_amdgcn_wmma_f32_16x16x32_f16(
                   false, ones, false, bp, (short)0, lc, false, false);
    }

    #pragma unroll
    for (int dt = 0; dt < 4; ++dt) {
        v16h a;  // A-fragment: V^T rows (M=d, K=key)
        const half_t* vr = &vTile[(dt * 16 + ln16) * VSTR + hh * 8];
        #pragma unroll
        for (int v = 0; v < 4; ++v) {
            h2 p0 = *(const h2*)(vr + 2 * v);
            h2 p1 = *(const h2*)(vr + 2 * v + 16);
            a[2 * v]     = p0[0];
            a[2 * v + 1] = p0[1];
            a[2 * v + 8] = p1[0];
            a[2 * v + 9] = p1[1];
        }
        v8f o = oacc[dt];
        #pragma unroll
        for (int r = 0; r < 8; ++r) o[r] *= alpha;
        oacc[dt] = __builtin_amdgcn_wmma_f32_16x16x32_f16(
                       false, a, false, bp, (short)0, o, false, false);
    }
}

__device__ __forceinline__
void fa_gload(const float* __restrict__ Kp, const float* __restrict__ Vp,
              int kb, int row, int dc,
              float4& kA, float4& kB, float4& vA, float4& vB)
{
    const float* ks = Kp + (size_t)(kb + row) * DH + dc;
    kA = *(const float4*)(ks);
    kB = *(const float4*)(ks + 4);
    const float* vs = Vp + (size_t)(kb + row) * DH + dc;
    vA = *(const float4*)(vs);
    vB = *(const float4*)(vs + 4);
}

__device__ __forceinline__
void fa_stage(half_t* __restrict__ kTile, half_t* __restrict__ vTile,
              int row, int dc, float4 kA, float4 kB, float4 vA, float4 vB)
{
    auto c0 = __builtin_amdgcn_cvt_pkrtz(kA.x, kA.y);
    auto c1 = __builtin_amdgcn_cvt_pkrtz(kA.z, kA.w);
    auto c2 = __builtin_amdgcn_cvt_pkrtz(kB.x, kB.y);
    auto c3 = __builtin_amdgcn_cvt_pkrtz(kB.z, kB.w);
    h8 hk;
    hk[0] = (_Float16)c0[0]; hk[1] = (_Float16)c0[1];
    hk[2] = (_Float16)c1[0]; hk[3] = (_Float16)c1[1];
    hk[4] = (_Float16)c2[0]; hk[5] = (_Float16)c2[1];
    hk[6] = (_Float16)c3[0]; hk[7] = (_Float16)c3[1];
    *(h8*)&kTile[row * KSTR + dc] = hk;

    vTile[(dc + 0) * VSTR + row] = (_Float16)vA.x;
    vTile[(dc + 1) * VSTR + row] = (_Float16)vA.y;
    vTile[(dc + 2) * VSTR + row] = (_Float16)vA.z;
    vTile[(dc + 3) * VSTR + row] = (_Float16)vA.w;
    vTile[(dc + 4) * VSTR + row] = (_Float16)vB.x;
    vTile[(dc + 5) * VSTR + row] = (_Float16)vB.y;
    vTile[(dc + 6) * VSTR + row] = (_Float16)vB.z;
    vTile[(dc + 7) * VSTR + row] = (_Float16)vB.w;
}

__global__ __launch_bounds__(256)
void fa_causal_wmma_kernel(const float* __restrict__ Qg,
                           const float* __restrict__ Kg,
                           const float* __restrict__ Vg,
                           float* __restrict__ Og)
{
    __shared__ __align__(16) half_t kTile[KB * KSTR];   // [key][d], f16
    __shared__ __align__(16) half_t vTile[DH * VSTR];   // [d][key], f16 (V transposed)

    const int bh   = blockIdx.y;
    const int q0wg = blockIdx.x * QWG;
    const int tid  = threadIdx.x;
    const int wave = tid >> 5;
    const int lane = tid & 31;
    const int ln16 = lane & 15;
    const int hh   = lane >> 4;          // half-wave (0/1)

    const size_t base = (size_t)bh * SEQ * DH;
    const float* Qp = Qg + base;
    const float* Kp = Kg + base;
    const float* Vp = Vg + base;
    float*       Op = Og + base;

    const int qlo  = q0wg + wave * QW;   // wave-uniform
    const int qhi  = qlo + QW - 1;       // wave-uniform
    const int qrow = qlo + ln16;         // this lane's query (column N of S^T)

    // ---- preload Q as two B-fragments, scale*log2e folded in before f16 ----
    v16h bq[2];
    #pragma unroll
    for (int c = 0; c < 2; ++c) {
        const float* qr = Qp + (size_t)qrow * DH + c * 32 + hh * 8;
        #pragma unroll
        for (int v = 0; v < 4; ++v) {
            float2 a = *(const float2*)(qr + 2 * v);
            float2 b = *(const float2*)(qr + 2 * v + 16);
            auto pa = __builtin_amdgcn_cvt_pkrtz(a.x * SCALE, a.y * SCALE);
            auto pb = __builtin_amdgcn_cvt_pkrtz(b.x * SCALE, b.y * SCALE);
            bq[c][2 * v]     = (_Float16)pa[0];
            bq[c][2 * v + 1] = (_Float16)pa[1];
            bq[c][2 * v + 8] = (_Float16)pb[0];
            bq[c][2 * v + 9] = (_Float16)pb[1];
        }
    }

    // ---- loop-invariant all-ones A-fragment (rowsum via matrix pipe) ----
    v16h ones;
    #pragma unroll
    for (int i = 0; i < 16; ++i) ones[i] = (_Float16)1.0f;

    // ---- flash state ----
    v8f oacc[4];
    #pragma unroll
    for (int t = 0; t < 4; ++t) oacc[t] = (v8f){};
    v8f lacc = (v8f){};
    float m = -__builtin_inff();

    const int nblk = (q0wg + QWG) / KB;  // causal frontier for this workgroup
    const int row  = tid >> 3;           // 0..31 (key within block), staging role
    const int dc   = (tid & 7) * 8;      // 0,8,...,56

    // ---- software pipeline: prefetch block j+1 into regs before compute(j) ----
    float4 kA, kB, vA, vB;
    fa_gload(Kp, Vp, 0, row, dc, kA, kB, vA, vB);

    for (int j = 0; j < nblk; ++j) {
        const int kb = j * KB;

        __syncthreads();                         // compute(j-1) done reading LDS
        fa_stage(kTile, vTile, row, dc, kA, kB, vA, vB);
        if (j + 1 < nblk)                        // issue next block's loads now;
            fa_gload(Kp, Vp, kb + KB, row, dc, kA, kB, vA, vB);  // they overlap compute(j)
        __syncthreads();                         // LDS block j visible

        if (kb + KB - 1 <= qlo) {                // wave-uniform: fully unmasked block
            fa_block<false>(kTile, vTile, kb, qrow, ln16, hh, bq, ones, oacc, lacc, m);
        } else if (kb <= qhi) {                  // wave-uniform: the one boundary block
            fa_block<true>(kTile, vTile, kb, qrow, ln16, hh, bq, ones, oacc, lacc, m);
        }
    }

    // ---- epilogue: every row of the ones-tile = rowsum(P) for this lane's q ----
    const float invl = 1.0f / lacc[0];
    #pragma unroll
    for (int dt = 0; dt < 4; ++dt) {
        float4 lo, hi;
        lo.x = oacc[dt][0] * invl; lo.y = oacc[dt][1] * invl;
        lo.z = oacc[dt][2] * invl; lo.w = oacc[dt][3] * invl;
        hi.x = oacc[dt][4] * invl; hi.y = oacc[dt][5] * invl;
        hi.z = oacc[dt][6] * invl; hi.w = oacc[dt][7] * invl;
        float* dst = Op + (size_t)qrow * DH + dt * 16 + hh * 8;
        *(float4*)(dst)     = lo;
        *(float4*)(dst + 4) = hi;
    }
}

extern "C" void kernel_launch(void* const* d_in, const int* in_sizes, int n_in,
                              void* d_out, int out_size, void* d_ws, size_t ws_size,
                              hipStream_t stream) {
    (void)in_sizes; (void)n_in; (void)out_size; (void)d_ws; (void)ws_size;
    const float* Q = (const float*)d_in[0];
    const float* K = (const float*)d_in[1];
    const float* V = (const float*)d_in[2];
    // d_in[3] is the causal bool mask; causality is computed analytically.
    float* O = (float*)d_out;

    dim3 grid(SEQ / QWG, NBH);   // (16, 64)
    dim3 block(256);             // 8 waves of 32
    hipLaunchKernelGGL(fa_causal_wmma_kernel, grid, block, 0, stream, Q, K, V, O);
}
